// LocalDecoder_12549894439062
// MI455X (gfx1250) — matrix operand order repaired and checked by reading, not compiled, not run on hardware
//
#include <hip/hip_runtime.h>
#include <hip/hip_bf16.h>
#include <cstdint>
#include <cstddef>

// ---------------- problem constants (from the reference) ----------------
#define LDEC_L  4
#define LDEC_D  512
#define LDEC_G  768
#define LDEC_HC 8
#define LDEC_HS 8
#define LDEC_F  2048
#define LDEC_B  8
#define LDEC_T  2048
#define LDEC_NP 256
#define HDIM    64      // head dim for both attentions (512/8)

typedef _Float16 h16;
typedef __attribute__((ext_vector_type(16))) _Float16 v16h;
typedef __attribute__((ext_vector_type(8)))  _Float16 v8h;
typedef __attribute__((ext_vector_type(8)))  float    v8f;
typedef unsigned int v4u __attribute__((ext_vector_type(4)));
typedef int          v4i __attribute__((ext_vector_type(4)));
typedef int          v8i_t __attribute__((ext_vector_type(8)));

#define WMMA16(a, b, c) __builtin_amdgcn_wmma_f32_16x16x32_f16( \
    false, (a), false, (b), (short)0, (c), false, false)

#if __has_builtin(__builtin_amdgcn_tensor_load_to_lds) && \
    __has_builtin(__builtin_amdgcn_s_wait_tensorcnt)
#define HAVE_TDM 1
#endif

// ---------------- fragment loaders (wave32 WMMA; contiguous k order, same in A and B) ----------------
// A (16x32 f16): lanes 0-15 row M=lane, K = 0..15;
//                lanes 16-31 same rows, K = 16..31.
__device__ __forceinline__ v16h load_a_frag(const h16* __restrict__ base, int ld,
                                            int m0, int kb, int lane) {
  int m  = m0 + (lane & 15);
  int ks = (lane >> 4) * 16;
  const h16* p = base + (size_t)m * ld + kb + ks;
  v8h lo = *(const v8h*)p;
  v8h hi = *(const v8h*)(p + 8);
  v16h a;
#pragma unroll
  for (int i = 0; i < 8; ++i) { a[i] = lo[i]; a[8 + i] = hi[i]; }
  return a;
}

// B (32x16 f16): lane holds column n=lane&15; K = (lane>>4)*16 + i (contiguous 16).
// For C = A x W^T with row-major W[N,K], lane n loads 16 contiguous halfs of W row n.
__device__ __forceinline__ v16h load_b_frag(const h16* __restrict__ base, int ld,
                                            int n0, int kb, int lane) {
  int n  = n0 + (lane & 15);
  int ks = (lane >> 4) * 16;
  return *(const v16h*)(base + (size_t)n * ld + kb + ks);
}

__device__ __forceinline__ float wave_sum32(float v) {
#pragma unroll
  for (int m = 1; m < 32; m <<= 1) v += __shfl_xor(v, m, 32);
  return v;
}

// ---------------- TDM: DMA a 64(N) x 32(K) f16 tile of W into LDS ----------------
// D# per CDNA5 ISA ch.8: group0 = {count/flags, lds_addr, global_addr, type=2},
// group1 = {data_size=2B, tensor dims, tile dims 32x64, dim0 stride = K halfs}.
__device__ __forceinline__ void tdm_load_tile_64x32(const h16* gsrc, int ldK, h16* ldst) {
#ifdef HAVE_TDM
  uint64_t ga = (uint64_t)(uintptr_t)gsrc;
  uint32_t lo = (uint32_t)(uintptr_t)ldst;   // flat LDS ptr low 32 bits = LDS byte offset
  v4u g0;
  g0[0] = 1u;                                              // count=1 (user descriptor)
  g0[1] = lo;                                              // lds_addr (bytes)
  g0[2] = (uint32_t)ga;                                    // global_addr[31:0]
  g0[3] = (uint32_t)((ga >> 32) & 0x01ffffffu) | (2u << 30); // global_addr[56:32] | type=2
  v8i_t g1;
  g1[0] = 0x00010000;                                      // data_size = 2 bytes
  g1[1] = (32 << 16);                                      // tensor_dim0 = 32 (halfs)
  g1[2] = (64 << 16);                                      // tensor_dim1 = 64 (rows)
  g1[3] = (32 << 16);                                      // tile_dim0 = 32
  g1[4] = 64;                                              // tile_dim1 = 64, tile_dim2 = 0
  g1[5] = ldK;                                             // tensor_dim0_stride = K (halfs)
  g1[6] = 0;
  g1[7] = 0;
  v4i z4 = {0, 0, 0, 0};
#if __clang_major__ >= 23
  v8i_t z8 = {0, 0, 0, 0, 0, 0, 0, 0};
  __builtin_amdgcn_tensor_load_to_lds(g0, g1, z4, z4, z8, 0);
#else
  __builtin_amdgcn_tensor_load_to_lds(g0, g1, z4, z4, 0);
#endif
#else
  (void)gsrc; (void)ldK; (void)ldst;
#endif
}

// ---------------- GEMM: C[M,N] = A[M,K] (f16) x W[N,K]^T (f16) + bias ----------------
// block = 256 threads = 8 waves; wave tile 32x64; block tile 256x64.
// Weight tile staged in LDS (TDM double-buffered; cooperative copy fallback).
template <int OUT_F16, int RELU>
__global__ __launch_bounds__(256)
void gemm_bias(const h16* __restrict__ A, const h16* __restrict__ W,
               const float* __restrict__ bias,
               float* __restrict__ Cf, h16* __restrict__ Ch,
               int M, int N, int K) {
  __shared__ __align__(16) h16 btile[2][64 * 32];
  int lane = threadIdx.x & 31;
  int wave = threadIdx.x >> 5;
  int m0 = blockIdx.y * 256 + wave * 32;
  int n0 = blockIdx.x * 64;
  (void)M;
  v8f acc[2][4] = {};
  const int nK = K / 32;
  for (int kt = 0; kt < nK; ++kt) {
    int kb = kt * 32;
#ifdef HAVE_TDM
    if (wave == 0) {
      if (kt == 0)
        tdm_load_tile_64x32(W + (size_t)n0 * K, K, &btile[0][0]);
      if (kt + 1 < nK) {
        tdm_load_tile_64x32(W + (size_t)n0 * K + kb + 32, K, &btile[(kt + 1) & 1][0]);
        __builtin_amdgcn_s_wait_tensorcnt(1);   // current tile complete (in-order TDM)
      } else {
        __builtin_amdgcn_s_wait_tensorcnt(0);
      }
    }
#else
    {
      int t = threadIdx.x;                      // 256 threads x 8 halfs = 64x32 tile
      int row = t >> 2, col = (t & 3) * 8;
      *(v8h*)(&btile[kt & 1][row * 32 + col]) =
          *(const v8h*)(W + (size_t)(n0 + row) * K + kb + col);
    }
#endif
    __syncthreads();
    const h16* bt = &btile[kt & 1][0];
    if (kb + 32 < K)
      __builtin_prefetch(A + (size_t)(m0 + (lane & 15)) * K + kb + 32, 0, 1);
    int ks = (lane >> 4) * 16;
    int nlo = lane & 15;
#pragma unroll
    for (int mt = 0; mt < 2; ++mt) {
      v16h a = load_a_frag(A, K, m0 + mt * 16, kb, lane);
#pragma unroll
      for (int nt = 0; nt < 4; ++nt) {
        v16h b = *(const v16h*)(bt + (nt * 16 + nlo) * 32 + ks);
        acc[mt][nt] = WMMA16(a, b, acc[mt][nt]);
      }
    }
    __syncthreads();
  }
  int hl  = lane >> 4;
  int nlo = lane & 15;
#pragma unroll
  for (int nt = 0; nt < 4; ++nt) {
    int n = n0 + nt * 16 + nlo;
    float bv = bias ? bias[n] : 0.0f;
#pragma unroll
    for (int mt = 0; mt < 2; ++mt) {
#pragma unroll
      for (int j = 0; j < 8; ++j) {
        int m = m0 + mt * 16 + j + 8 * hl;       // C row = j + 8*(lane>>4)
        float v = acc[mt][nt][j] + bv;
        if (RELU) v = fmaxf(v, 0.0f);
        if (OUT_F16) Ch[(size_t)m * N + n] = (h16)v;
        else         Cf[(size_t)m * N + n] = v;
      }
    }
  }
}

// ---------------- LayerNorm (wave per row), f16 output ----------------
__global__ __launch_bounds__(256)
void ln_f16(const float* __restrict__ x, const float* __restrict__ g,
            const float* __restrict__ b, h16* __restrict__ out, int M, int Dw) {
  int row = blockIdx.x * 8 + (threadIdx.x >> 5);
  if (row >= M) return;
  int lane = threadIdx.x & 31;
  const float* xr = x + (size_t)row * Dw;
  float s = 0.0f;
  for (int i = lane; i < Dw; i += 32) s += xr[i];
  float mean = wave_sum32(s) / (float)Dw;
  float v = 0.0f;
  for (int i = lane; i < Dw; i += 32) { float d = xr[i] - mean; v += d * d; }
  float inv = rsqrtf(wave_sum32(v) / (float)Dw + 1e-5f);
  for (int i = lane; i < Dw; i += 32)
    out[(size_t)row * Dw + i] = (h16)((xr[i] - mean) * inv * g[i] + b[i]);
}

// ---------------- y = LN(x + delta); writes fp32 (residual stream) + f16 copy ----------------
__global__ __launch_bounds__(256)
void add_ln(const float* __restrict__ x, const float* __restrict__ dl,
            const float* __restrict__ g, const float* __restrict__ bb,
            float* __restrict__ xo, h16* __restrict__ xh, int M) {
  int row = blockIdx.x * 8 + (threadIdx.x >> 5);
  if (row >= M) return;
  int lane = threadIdx.x & 31;
  const int Dw = LDEC_D;
  float yv[Dw / 32];
  float s = 0.0f;
#pragma unroll
  for (int i = 0; i < Dw / 32; ++i) {
    int c = lane + 32 * i;
    float y = x[(size_t)row * Dw + c] + dl[(size_t)row * Dw + c];
    yv[i] = y; s += y;
  }
  float mean = wave_sum32(s) * (1.0f / Dw);
  float v = 0.0f;
#pragma unroll
  for (int i = 0; i < Dw / 32; ++i) { float d = yv[i] - mean; v += d * d; }
  float inv = rsqrtf(wave_sum32(v) * (1.0f / Dw) + 1e-5f);
#pragma unroll
  for (int i = 0; i < Dw / 32; ++i) {
    int c = lane + 32 * i;
    float o = (yv[i] - mean) * inv * g[c] + bb[c];
    xo[(size_t)row * Dw + c] = o;
    xh[(size_t)row * Dw + c] = (h16)o;
  }
}

// ---------------- streaming-softmax attention (cross & self) ----------------
// ctx[b,t,h*64+d] = softmax_k( Q[t]·K[k] / 8 + mask ) · V[k]
// Q rows at Qp + (b*T + t)*ldq + h*64 ; K rows at Kp + (b*nk + k)*ldk + h*64
// Vt is V transposed per head: [B, H, 64, nk]  (rows contiguous over keys)
__global__ __launch_bounds__(256)
void attn_kernel(const h16* __restrict__ Qp, int ldq,
                 const h16* __restrict__ Kp, int ldk,
                 const h16* __restrict__ Vt,
                 h16* __restrict__ ctx,
                 const int* __restrict__ pids,   // null => no mask (self-attn)
                 int nk, int H) {
  __shared__ h16 ptile[8 * 16 * 32];            // per-wave 16x32 P tile (f16)
  int lane = threadIdx.x & 31;
  int wave = threadIdx.x >> 5;
  int bh = blockIdx.x;
  int b = bh / H, h = bh % H;
  int t0 = blockIdx.y * 128 + wave * 16;
  int hl = lane >> 4, nlo = lane & 15;
  const float SCALE = 0.125f;                   // 1/sqrt(64)

  const h16* qbase = Qp + (size_t)b * LDEC_T * ldq + h * HDIM;
  const h16* kbase = Kp + (size_t)b * nk * ldk + h * HDIM;
  const h16* vbase = Vt + ((size_t)(b * H + h) * HDIM) * nk;

  v16h aq0 = load_a_frag(qbase, ldq, t0, 0, lane);
  v16h aq1 = load_a_frag(qbase, ldq, t0, 32, lane);

  int pid[8];
  if (pids) {
#pragma unroll
    for (int j = 0; j < 8; ++j)
      pid[j] = pids[(size_t)b * LDEC_T + t0 + j + 8 * hl];
  }

  float mrun[8], srun[8], fac[8];
#pragma unroll
  for (int j = 0; j < 8; ++j) { mrun[j] = -1e30f; srun[j] = 0.0f; }
  v8f oc[4] = {};

  h16* myp = &ptile[wave * 512];

  for (int p0 = 0; p0 < nk; p0 += 32) {
    // ---- S = Q K^T for 16 rows x 32 keys (2 col-fragments, K-dim 64 = 2 chunks)
    v8f s0 = {}, s1 = {};
#pragma unroll
    for (int ch = 0; ch < 2; ++ch) {
      v16h aq = (ch == 0) ? aq0 : aq1;
      v16h bk0 = load_b_frag(kbase, ldk, p0, ch * 32, lane);
      v16h bk1 = load_b_frag(kbase, ldk, p0 + 16, ch * 32, lane);
      s0 = WMMA16(aq, bk0, s0);
      s1 = WMMA16(aq, bk1, s1);
    }
    // ---- online softmax update (rows spread over 16-lane half-groups)
#pragma unroll
    for (int j = 0; j < 8; ++j) {
      float e0 = s0[j] * SCALE;
      float e1 = s1[j] * SCALE;
      if (pids) {
        if (p0 + nlo      > pid[j]) e0 = -1e9f;
        if (p0 + 16 + nlo > pid[j]) e1 = -1e9f;
      }
      float mx = fmaxf(e0, e1);
#pragma unroll
      for (int msk = 1; msk < 16; msk <<= 1) mx = fmaxf(mx, __shfl_xor(mx, msk, 32));
      float mnew = fmaxf(mrun[j], mx);
      float f = __expf(mrun[j] - mnew);
      float pv0 = __expf(e0 - mnew);
      float pv1 = __expf(e1 - mnew);
      float rs = pv0 + pv1;
#pragma unroll
      for (int msk = 1; msk < 16; msk <<= 1) rs += __shfl_xor(rs, msk, 32);
      srun[j] = srun[j] * f + rs;
      mrun[j] = mnew;
      fac[j] = f;
      int r = j + 8 * hl;
      myp[r * 32 + nlo]      = (h16)pv0;
      myp[r * 32 + 16 + nlo] = (h16)pv1;
    }
#pragma unroll
    for (int nt = 0; nt < 4; ++nt)
#pragma unroll
      for (int j = 0; j < 8; ++j) oc[nt][j] *= fac[j];
    __syncthreads();
    // ---- reload P in A-fragment layout from LDS, accumulate O += P * V
    {
      int m = nlo, ks = hl * 16;
      v8h lo = *(const v8h*)(myp + m * 32 + ks);
      v8h hi = *(const v8h*)(myp + m * 32 + 8 + ks);
      v16h ap;
#pragma unroll
      for (int i = 0; i < 8; ++i) { ap[i] = lo[i]; ap[8 + i] = hi[i]; }
#pragma unroll
      for (int nt = 0; nt < 4; ++nt) {
        v16h bv = load_b_frag(vbase, nk, nt * 16, p0, lane);
        oc[nt] = WMMA16(ap, bv, oc[nt]);
      }
    }
    __syncthreads();
  }
  // ---- normalize and store f16 context
#pragma unroll
  for (int j = 0; j < 8; ++j) {
    float inv = 1.0f / srun[j];
    int t = t0 + j + 8 * hl;
    size_t ro = ((size_t)b * LDEC_T + t) * (size_t)(H * HDIM) + h * HDIM;
#pragma unroll
    for (int nt = 0; nt < 4; ++nt)
      ctx[ro + nt * 16 + nlo] = (h16)(oc[nt][j] * inv);
  }
}

// ---------------- V -> V^T per head: Vt[b,h,d,p] = V[(b*nk+p)*ld + h*64 + d] ----------------
__global__ void transpose_v(const h16* __restrict__ V, int ld,
                            h16* __restrict__ Vt, int nk, int H, size_t total) {
  size_t idx = (size_t)blockIdx.x * blockDim.x + threadIdx.x;
  if (idx >= total) return;
  int p = (int)(idx % nk);
  size_t r = idx / nk;
  int dd = (int)(r % HDIM); r /= HDIM;
  int h = (int)(r % H);
  int b = (int)(r / H);
  Vt[idx] = V[((size_t)b * nk + p) * ld + h * HDIM + dd];
}

__global__ void f32_to_f16(const float* __restrict__ in, h16* __restrict__ out, size_t n) {
  size_t i = (size_t)blockIdx.x * blockDim.x + threadIdx.x;
  if (i < n) out[i] = (h16)in[i];
}

// patch_ids[b,t] = cumsum(boundaries[b,:t+1]) - 1
__global__ void compute_pids(const int* __restrict__ bounds, int* __restrict__ pids, int Tn) {
  if (threadIdx.x != 0) return;
  int b = blockIdx.x;
  int c = 0;
  for (int t = 0; t < Tn; ++t) {
    c += bounds[(size_t)b * Tn + t];
    pids[(size_t)b * Tn + t] = c - 1;
  }
}

// ---------------- host driver ----------------
extern "C" void kernel_launch(void* const* d_in, const int* in_sizes, int n_in,
                              void* d_out, int out_size, void* d_ws, size_t ws_size,
                              hipStream_t stream) {
  (void)in_sizes; (void)n_in; (void)out_size; (void)ws_size;
  const int D = LDEC_D, G = LDEC_G, T = LDEC_T, NP = LDEC_NP, F = LDEC_F;
  const int Bn = LDEC_B, Ln = LDEC_L, HC = LDEC_HC, HS = LDEC_HS;
  const int Mx = Bn * T;     // 16384
  const int Mkv = Bn * NP;   // 2048

  const float* patch_reprs = (const float*)d_in[0];
  const float* enc         = (const float*)d_in[1];
  const int*   bounds      = (const int*)d_in[2];
  const float* ca_qw = (const float*)d_in[3];
  const float* ca_qb = (const float*)d_in[4];
  const float* ca_kw = (const float*)d_in[5];
  const float* ca_kb = (const float*)d_in[6];
  const float* ca_vw = (const float*)d_in[7];
  const float* ca_vb = (const float*)d_in[8];
  const float* ca_ow = (const float*)d_in[9];
  const float* ca_ob = (const float*)d_in[10];
  const float* nq_g  = (const float*)d_in[11];
  const float* nq_b  = (const float*)d_in[12];
  const float* nkv_g = (const float*)d_in[13];
  const float* nkv_b = (const float*)d_in[14];
  const float* ln_g  = (const float*)d_in[15];
  const float* ln_b  = (const float*)d_in[16];
  const float* sa_inw = (const float*)d_in[17];
  const float* sa_inb = (const float*)d_in[18];
  const float* sa_ow  = (const float*)d_in[19];
  const float* sa_ob  = (const float*)d_in[20];
  const float* l1w = (const float*)d_in[21];
  const float* l1b = (const float*)d_in[22];
  const float* l2w = (const float*)d_in[23];
  const float* l2b = (const float*)d_in[24];
  const float* n1_g = (const float*)d_in[25];
  const float* n1_b = (const float*)d_in[26];
  const float* n2_g = (const float*)d_in[27];
  const float* n2_b = (const float*)d_in[28];
  const float* out_w = (const float*)d_in[29];
  const float* out_b = (const float*)d_in[30];

  char* base = (char*)d_ws;
  size_t off = 0;
  auto alloc = [&](size_t bytes) -> void* {
    void* p = base + off;
    off = (off + bytes + 255) & ~(size_t)255;
    return p;
  };
  h16* wq16  = (h16*)alloc((size_t)Ln * D * D * 2);
  h16* wk16  = (h16*)alloc((size_t)Ln * D * G * 2);
  h16* wv16  = (h16*)alloc((size_t)Ln * D * G * 2);
  h16* wo16  = (h16*)alloc((size_t)Ln * D * D * 2);
  h16* win16 = (h16*)alloc((size_t)Ln * 3 * D * D * 2);
  h16* wso16 = (h16*)alloc((size_t)Ln * D * D * 2);
  h16* w116  = (h16*)alloc((size_t)Ln * F * D * 2);
  h16* w216  = (h16*)alloc((size_t)Ln * D * F * 2);
  h16* wout16 = (h16*)alloc((size_t)256 * D * 2);
  float* x  = (float*)alloc((size_t)Mx * D * 4);
  float* cf = (float*)alloc((size_t)Mx * D * 4);
  h16* xh   = (h16*)alloc((size_t)Mx * D * 2);
  h16* qn   = (h16*)alloc((size_t)Mx * D * 2);
  h16* kvn  = (h16*)alloc((size_t)Mkv * G * 2);
  h16* Qh   = (h16*)alloc((size_t)Mx * D * 2);
  h16* Kh   = (h16*)alloc((size_t)Mkv * D * 2);
  h16* Vh   = (h16*)alloc((size_t)Mkv * D * 2);
  h16* Vtc  = (h16*)alloc((size_t)Mkv * D * 2);
  h16* ctx  = (h16*)alloc((size_t)Mx * D * 2);
  h16* qkvh = (h16*)alloc((size_t)Mx * 3 * D * 2);
  h16* Vst  = (h16*)alloc((size_t)Mx * D * 2);
  h16* h1h  = (h16*)alloc((size_t)Mx * F * 2);
  int* pids = (int*)alloc((size_t)Bn * T * 4);

  auto cvt = [&](const float* s, h16* d, size_t n) {
    f32_to_f16<<<dim3((unsigned)((n + 255) / 256)), 256, 0, stream>>>(s, d, n);
  };
  cvt(ca_qw,  wq16,  (size_t)Ln * D * D);
  cvt(ca_kw,  wk16,  (size_t)Ln * D * G);
  cvt(ca_vw,  wv16,  (size_t)Ln * D * G);
  cvt(ca_ow,  wo16,  (size_t)Ln * D * D);
  cvt(sa_inw, win16, (size_t)Ln * 3 * D * D);
  cvt(sa_ow,  wso16, (size_t)Ln * D * D);
  cvt(l1w,    w116,  (size_t)Ln * F * D);
  cvt(l2w,    w216,  (size_t)Ln * D * F);
  cvt(out_w,  wout16, (size_t)256 * D);
  hipMemcpyAsync(x, enc, (size_t)Mx * D * 4, hipMemcpyDeviceToDevice, stream);
  compute_pids<<<Bn, 32, 0, stream>>>(bounds, pids, T);

  for (int l = 0; l < Ln; ++l) {
    // ---- cross attention ----
    ln_f16<<<Mx / 8, 256, 0, stream>>>(x, nq_g + (size_t)l * D, nq_b + (size_t)l * D, qn, Mx, D);
    ln_f16<<<Mkv / 8, 256, 0, stream>>>(patch_reprs, nkv_g + (size_t)l * G, nkv_b + (size_t)l * G, kvn, Mkv, G);
    gemm_bias<1, 0><<<dim3(D / 64, Mx / 256), 256, 0, stream>>>(
        qn, wq16 + (size_t)l * D * D, ca_qb + (size_t)l * D, nullptr, Qh, Mx, D, D);
    gemm_bias<1, 0><<<dim3(D / 64, Mkv / 256), 256, 0, stream>>>(
        kvn, wk16 + (size_t)l * D * G, ca_kb + (size_t)l * D, nullptr, Kh, Mkv, D, G);
    gemm_bias<1, 0><<<dim3(D / 64, Mkv / 256), 256, 0, stream>>>(
        kvn, wv16 + (size_t)l * D * G, ca_vb + (size_t)l * D, nullptr, Vh, Mkv, D, G);
    {
      size_t tot = (size_t)Bn * HC * HDIM * NP;
      transpose_v<<<dim3((unsigned)((tot + 255) / 256)), 256, 0, stream>>>(Vh, D, Vtc, NP, HC, tot);
    }
    attn_kernel<<<dim3(Bn * HC, T / 128), 256, 0, stream>>>(Qh, D, Kh, D, Vtc, ctx, pids, NP, HC);
    gemm_bias<0, 0><<<dim3(D / 64, Mx / 256), 256, 0, stream>>>(
        ctx, wo16 + (size_t)l * D * D, ca_ob + (size_t)l * D, cf, nullptr, Mx, D, D);
    add_ln<<<Mx / 8, 256, 0, stream>>>(x, cf, ln_g + (size_t)l * D, ln_b + (size_t)l * D, x, xh, Mx);
    // ---- self attention ----
    gemm_bias<1, 0><<<dim3(3 * D / 64, Mx / 256), 256, 0, stream>>>(
        xh, win16 + (size_t)l * 3 * D * D, sa_inb + (size_t)l * 3 * D, nullptr, qkvh, Mx, 3 * D, D);
    {
      size_t tot = (size_t)Bn * HS * HDIM * T;
      transpose_v<<<dim3((unsigned)((tot + 255) / 256)), 256, 0, stream>>>(qkvh + 2 * D, 3 * D, Vst, T, HS, tot);
    }
    attn_kernel<<<dim3(Bn * HS, T / 128), 256, 0, stream>>>(qkvh, 3 * D, qkvh + D, 3 * D, Vst, ctx, nullptr, T, HS);
    gemm_bias<0, 0><<<dim3(D / 64, Mx / 256), 256, 0, stream>>>(
        ctx, wso16 + (size_t)l * D * D, sa_ob + (size_t)l * D, cf, nullptr, Mx, D, D);
    add_ln<<<Mx / 8, 256, 0, stream>>>(x, cf, n1_g + (size_t)l * D, n1_b + (size_t)l * D, x, xh, Mx);
    // ---- FFN ----
    gemm_bias<1, 1><<<dim3(F / 64, Mx / 256), 256, 0, stream>>>(
        xh, w116 + (size_t)l * F * D, l1b + (size_t)l * F, nullptr, h1h, Mx, F, D);
    gemm_bias<0, 0><<<dim3(D / 64, Mx / 256), 256, 0, stream>>>(
        h1h, w216 + (size_t)l * D * F, l2b + (size_t)l * D, cf, nullptr, Mx, D, F);
    add_ln<<<Mx / 8, 256, 0, stream>>>(x, cf, n2_g + (size_t)l * D, n2_b + (size_t)l * D, x, xh, Mx);
  }
  // ---- output head: [Mx,512] x [256,512]^T + out_b -> d_out (fp32) ----
  gemm_bias<0, 0><<<dim3(256 / 64, Mx / 256), 256, 0, stream>>>(
      xh, wout16, out_b, (float*)d_out, nullptr, Mx, 256, D);
}
